// Attention_10771777978397
// MI455X (gfx1250) — matrix operand order repaired and checked by reading, not compiled, not run on hardware
//
#include <hip/hip_runtime.h>
#include <hip/hip_bf16.h>
#include <math.h>

#define B_  4
#define S_  1024
#define D_  2048
#define H_  16
#define HD_ 128

typedef _Float16 h16v __attribute__((ext_vector_type(16)));
typedef _Float16 h8v  __attribute__((ext_vector_type(8)));
typedef float    f8v  __attribute__((ext_vector_type(8)));

__device__ __forceinline__ f8v wmma32(h16v a, h16v b, f8v c) {
  // D = A(16x32 f16) x B(32x16 f16) + C(16x16 f32)
  return __builtin_amdgcn_wmma_f32_16x16x32_f16(
      /*neg_a=*/false, a, /*neg_b=*/false, b,
      /*c_mod=*/(short)0, c, /*reuse_a=*/false, /*reuse_b=*/false);
}

__device__ __forceinline__ h16v make_a_frag(const _Float16* p) {
  // ISA A-fragment: lane holds row (lane&15), K-halves {c0..c0+7, c0+16..c0+23},
  // c0 = (lane<16 ? 0 : 8) folded into p by the caller. Two 16B loads.
  h8v lo = *(const h8v*)p;
  h8v hi = *(const h8v*)(p + 16);
  h16v a;
#pragma unroll
  for (int i = 0; i < 8; ++i) { a[i] = lo[i]; a[i + 8] = hi[i]; }
  return a;
}

// ---------------------------------------------------------------- converts
__global__ void cvt_f32_to_f16(const float* __restrict__ in,
                               _Float16* __restrict__ out, int n) {
  int i = blockIdx.x * blockDim.x + threadIdx.x;
  int stride = gridDim.x * blockDim.x;
  for (; i < n; i += stride) out[i] = (_Float16)in[i];
}

// Wt[k*N + n] = (f16) W[n*K + k]   (W is (out=N, in=K) row-major, torch Linear)
__global__ void transpose_to_f16(const float* __restrict__ W,
                                 _Float16* __restrict__ Wt, int N, int K) {
  int i = blockIdx.x * blockDim.x + threadIdx.x;
  int stride = gridDim.x * blockDim.x;
  int total = N * K;
  for (; i < total; i += stride) {
    int k = i / N, n = i - k * N;
    Wt[i] = (_Float16)W[(long)n * K + k];
  }
}

// ---------------------------------------------------------------- GEMM
// C[M,N] = A[M,K](f16) * Bt[K,N](f16). One wave computes a 16x64 strip:
// one A-fragment is reused across 4 B-fragments -> 4 WMMAs per K-step.
__global__ void __launch_bounds__(256)
gemm_wmma(const _Float16* __restrict__ A, const _Float16* __restrict__ Bt,
          _Float16* __restrict__ Ch, float* __restrict__ Cf,
          int M, int N, int K) {
  const int lane = threadIdx.x & 31;
  const int wave = threadIdx.x >> 5;
  const int tile = blockIdx.x * 8 + wave;
  const int tilesN = N >> 6;                       // 64-wide strips
  if (tile >= (M >> 4) * tilesN) return;
  const int m0 = (tile / tilesN) << 4;
  const int n0 = (tile - (tile / tilesN) * tilesN) << 6;

  const int am = m0 + (lane & 15);
  const int ac = (lane < 16) ? 0 : 8;              // A-fragment half-wave K offset
  const _Float16* arow = A + (long)am * K + ac;
  const _Float16* brow = Bt + (long)lane * N + n0; // B row K = lane (+kk)

  f8v acc0 = {}, acc1 = {}, acc2 = {}, acc3 = {};
  for (int kk = 0; kk < K; kk += 32) {
    if (kk + 32 < K) {
      __builtin_prefetch(arow + kk + 32, 0, 1);                  // global_prefetch_b8
      __builtin_prefetch(brow + (long)(kk + 32) * N, 0, 1);
      __builtin_prefetch(brow + (long)(kk + 32) * N + 32, 0, 1);
    }
    h16v a = make_a_frag(arow + kk);
    const _Float16* bp = Bt + (long)(kk + ac) * N + n0 + (lane & 15);
    h16v b0, b1, b2, b3;
#pragma unroll
    for (int i = 0; i < 8; ++i) {
      const _Float16* q0 = bp + (long)i * N;
      const _Float16* q1 = bp + (long)(i + 16) * N;
      b0[i] = q0[0];  b0[i + 8] = q1[0];
      b1[i] = q0[16]; b1[i + 8] = q1[16];
      b2[i] = q0[32]; b2[i + 8] = q1[32];
      b3[i] = q0[48]; b3[i + 8] = q1[48];
    }
    acc0 = wmma32(a, b0, acc0);
    acc1 = wmma32(a, b1, acc1);
    acc2 = wmma32(a, b2, acc2);
    acc3 = wmma32(a, b3, acc3);
  }
  const int rr0 = (lane < 16) ? 0 : 8;
  const int col = lane & 15;
  f8v accs[4] = {acc0, acc1, acc2, acc3};
  if (Cf) {
#pragma unroll
    for (int t = 0; t < 4; ++t)
#pragma unroll
      for (int r = 0; r < 8; ++r)
        Cf[(long)(m0 + rr0 + r) * N + n0 + t * 16 + col] = accs[t][r];
  } else {
#pragma unroll
    for (int t = 0; t < 4; ++t)
#pragma unroll
      for (int r = 0; r < 8; ++r)
        Ch[(long)(m0 + rr0 + r) * N + n0 + t * 16 + col] = (_Float16)accs[t][r];
  }
}

// ---------------------------------------------------------------- RoPE + layout
// qraw/kraw/vraw: [B,S,H*HD]. Writes qr:[B,H,S,HD], kt:[B,H,HD,S], v:[B,H,S,HD].
__global__ void rope_transpose(const _Float16* __restrict__ qraw,
                               const _Float16* __restrict__ kraw,
                               const _Float16* __restrict__ vraw,
                               const float* __restrict__ freqs,
                               _Float16* __restrict__ qr,
                               _Float16* __restrict__ kt,
                               _Float16* __restrict__ v) {
  const int total = B_ * H_ * S_ * HD_;
  int i = blockIdx.x * blockDim.x + threadIdx.x;
  const int stride = gridDim.x * blockDim.x;
  for (; i < total; i += stride) {
    int hd = i & (HD_ - 1);
    int s  = (i >> 7) & (S_ - 1);
    int h  = (i >> 17) & (H_ - 1);
    int b  = i >> 21;
    long rbase = (((long)b * S_ + s) * H_ + h) * HD_;
    int j = hd >> 1;
    float f = freqs[s * (HD_ / 2) + j];
    float c = cosf(f), sn = sinf(f);
    float qa = (float)qraw[rbase + (j << 1)];
    float qb = (float)qraw[rbase + (j << 1) + 1];
    float ka = (float)kraw[rbase + (j << 1)];
    float kb = (float)kraw[rbase + (j << 1) + 1];
    float qv = (hd & 1) ? (qa * sn + qb * c) : (qa * c - qb * sn);
    float kv = (hd & 1) ? (ka * sn + kb * c) : (ka * c - kb * sn);
    qr[i] = (_Float16)qv;
    kt[(((long)b * H_ + h) * HD_ + hd) * S_ + s] = (_Float16)kv;
    v[i] = vraw[rbase + hd];
  }
}

// ---------------------------------------------------------------- flash attention
// One wave per (b, h, 16-row q-tile). Online softmax over 32-key chunks.
__global__ void __launch_bounds__(256)
flash_attn(const _Float16* __restrict__ Q,   // [B,H,S,HD]
           const _Float16* __restrict__ Kt,  // [B,H,HD,S]
           const _Float16* __restrict__ V,   // [B,H,S,HD]
           _Float16* __restrict__ O)         // [B,S,H*HD]
{
  __shared__ __align__(64) _Float16 plds[8][16 * 32];  // per-wave 16x32 P tile
  const int lane = threadIdx.x & 31;
  const int wave = threadIdx.x >> 5;
  const int gw = blockIdx.x * 8 + wave;
  const int qt = gw & 63;
  const int h  = (gw >> 6) & 15;
  const int b  = gw >> 10;
  const int qb = qt << 4;

  const _Float16* qbase = Q + ((long)b * H_ + h) * S_ * HD_;
  const _Float16* kbase = Kt + ((long)b * H_ + h) * HD_ * S_;
  const _Float16* vbase = V + ((long)b * H_ + h) * S_ * HD_;

  const int am  = qb + (lane & 15);
  const int ac  = (lane < 16) ? 0 : 8;
  const int rr0 = (lane < 16) ? 0 : 8;
  const int col = lane & 15;

  // Resident Q fragments for the 4 HD chunks of 32
  h16v qf[4];
#pragma unroll
  for (int kc = 0; kc < 4; ++kc)
    qf[kc] = make_a_frag(qbase + (long)am * HD_ + kc * 32 + ac);

  f8v o[8];
#pragma unroll
  for (int t = 0; t < 8; ++t) o[t] = (f8v){};
  float mrow[8], lrow[8];
#pragma unroll
  for (int r = 0; r < 8; ++r) { mrow[r] = -3.0e38f; lrow[r] = 0.f; }

  const float scale = 0.08838834764831845f;  // 1/sqrt(128)

  for (int kb = 0; kb < qb + 16; kb += 32) {
    // scores: two 16x16 tiles (keys kb..+15, kb+16..+31)
    f8v s0 = {}, s1 = {};
#pragma unroll
    for (int kc = 0; kc < 4; ++kc) {
      const _Float16* kp = kbase + (long)(kc * 32 + ac) * S_ + kb + col;
      h16v b0, b1;
#pragma unroll
      for (int i = 0; i < 8; ++i) {
        b0[i] = kp[(long)i * S_];        b0[i + 8] = kp[(long)(i + 16) * S_];
        b1[i] = kp[(long)i * S_ + 16];   b1[i + 8] = kp[(long)(i + 16) * S_ + 16];
      }
      s0 = wmma32(qf[kc], b0, s0);
      s1 = wmma32(qf[kc], b1, s1);
    }
    // scale + causal mask + per-row tile max
    float pm[8];
#pragma unroll
    for (int r = 0; r < 8; ++r) {
      int qrow = qb + rr0 + r;
      float v0 = s0[r] * scale + ((kb + col      > qrow) ? -1e9f : 0.f);
      float v1 = s1[r] * scale + ((kb + 16 + col > qrow) ? -1e9f : 0.f);
      s0[r] = v0; s1[r] = v1;
      pm[r] = fmaxf(v0, v1);
    }
#pragma unroll
    for (int r = 0; r < 8; ++r) {
      float t = pm[r];
      t = fmaxf(t, __shfl_xor(t, 1, 32));
      t = fmaxf(t, __shfl_xor(t, 2, 32));
      t = fmaxf(t, __shfl_xor(t, 4, 32));
      t = fmaxf(t, __shfl_xor(t, 8, 32));   // stays within 16-lane row group
      pm[r] = t;
    }
#pragma unroll
    for (int r = 0; r < 8; ++r) {
      float mnew = fmaxf(mrow[r], pm[r]);
      float sf = __expf(mrow[r] - mnew);
      float p0 = __expf(s0[r] - mnew);
      float p1 = __expf(s1[r] - mnew);
      float ps = p0 + p1;
      ps += __shfl_xor(ps, 1, 32);
      ps += __shfl_xor(ps, 2, 32);
      ps += __shfl_xor(ps, 4, 32);
      ps += __shfl_xor(ps, 8, 32);
      lrow[r] = lrow[r] * sf + ps;
      mrow[r] = mnew;
#pragma unroll
      for (int t2 = 0; t2 < 8; ++t2) o[t2][r] *= sf;
      plds[wave][(rr0 + r) * 32 + col]      = (_Float16)p0;
      plds[wave][(rr0 + r) * 32 + 16 + col] = (_Float16)p1;
    }
    asm volatile("s_wait_dscnt 0" ::: "memory");  // wave-private LDS RAW fence
    // reload P (16x32) as an A-fragment
    h16v pf = make_a_frag(&plds[wave][(lane & 15) * 32 + ac]);
    // O += P * V   (8 HD tiles of 16)
#pragma unroll
    for (int t2 = 0; t2 < 8; ++t2) {
      const _Float16* vp = vbase + (long)(kb + ac) * HD_ + t2 * 16 + col;
      h16v bv;
#pragma unroll
      for (int i = 0; i < 8; ++i) { bv[i] = vp[i * HD_]; bv[i + 8] = vp[(i + 16) * HD_]; }
      o[t2] = wmma32(pf, bv, o[t2]);
    }
  }
  // epilogue: normalize, write [B,S,H*HD]
#pragma unroll
  for (int r = 0; r < 8; ++r) {
    float inv = 1.0f / lrow[r];
    int qrow = qb + rr0 + r;
    _Float16* op = O + ((long)b * S_ + qrow) * D_ + h * HD_;
#pragma unroll
    for (int t2 = 0; t2 < 8; ++t2)
      op[t2 * 16 + col] = (_Float16)(o[t2][r] * inv);
  }
}

// ---------------------------------------------------------------- launch
extern "C" void kernel_launch(void* const* d_in, const int* in_sizes, int n_in,
                              void* d_out, int out_size, void* d_ws, size_t ws_size,
                              hipStream_t stream) {
  (void)in_sizes; (void)n_in; (void)out_size; (void)ws_size;
  const float* x     = (const float*)d_in[0];
  const float* freqs = (const float*)d_in[1];
  // d_in[2] (mask) unused: causal mask applied analytically (matches reference)
  const float* wq = (const float*)d_in[3];
  const float* wk = (const float*)d_in[4];
  const float* wv = (const float*)d_in[5];
  const float* wo = (const float*)d_in[6];
  // d_in[7] (start_pos) == 0
  float* out = (float*)d_out;

  const long NX = (long)B_ * S_ * D_;  // 8,388,608
  const long NW = (long)D_ * D_;       // 4,194,304
  _Float16* ws = (_Float16*)d_ws;
  _Float16* xh   = ws; ws += NX;
  _Float16* wqt  = ws; ws += NW;
  _Float16* wkt  = ws; ws += NW;
  _Float16* wvt  = ws; ws += NW;
  _Float16* wot  = ws; ws += NW;
  _Float16* qraw = ws; ws += NX;
  _Float16* kraw = ws; ws += NX;
  _Float16* vraw = ws; ws += NX;
  _Float16* qr   = ws; ws += NX;
  _Float16* kt   = ws; ws += NX;
  _Float16* vt   = ws; ws += NX;
  _Float16* ao   = ws; ws += NX;

  cvt_f32_to_f16<<<2048, 256, 0, stream>>>(x, xh, (int)NX);
  transpose_to_f16<<<2048, 256, 0, stream>>>(wq, wqt, D_, D_);
  transpose_to_f16<<<2048, 256, 0, stream>>>(wk, wkt, D_, D_);
  transpose_to_f16<<<2048, 256, 0, stream>>>(wv, wvt, D_, D_);
  transpose_to_f16<<<2048, 256, 0, stream>>>(wo, wot, D_, D_);

  const int strips = (B_ * S_ / 16) * (D_ / 64);  // 8192 16x64 strips
  const int gblocks = strips / 8;                 // 8 waves per block
  gemm_wmma<<<gblocks, 256, 0, stream>>>(xh, wqt, qraw, nullptr, B_ * S_, D_, D_);
  gemm_wmma<<<gblocks, 256, 0, stream>>>(xh, wkt, kraw, nullptr, B_ * S_, D_, D_);
  gemm_wmma<<<gblocks, 256, 0, stream>>>(xh, wvt, vraw, nullptr, B_ * S_, D_, D_);

  rope_transpose<<<2048, 256, 0, stream>>>(qraw, kraw, vraw, freqs, qr, kt, vt);

  flash_attn<<<(B_ * H_ * (S_ / 16)) / 8, 256, 0, stream>>>(qr, kt, vt, ao);

  gemm_wmma<<<gblocks, 256, 0, stream>>>(ao, wot, nullptr, out, B_ * S_, D_, D_);
}
